// MultiScaleRetention_48988396978247
// MI455X (gfx1250) — matrix operand order repaired and checked by reading, not compiled, not run on hardware
//
#include <hip/hip_runtime.h>
#include <cstdint>

#define B_   16
#define S_   4096
#define DIN  1024
#define HD_  64
#define V_   512
#define FOUT 120
#define SCHUNK 32
#define LDSPITCH 1032   // bf16 elements per LDS row (1024 + 8 pad -> conflict-free)

typedef __attribute__((ext_vector_type(16))) __bf16 v16bf;
typedef __attribute__((ext_vector_type(8)))  float  v8f;

union AFrag { v16bf v; uint4 q[2]; };

__device__ __forceinline__ unsigned short f2bf(float f) {
  unsigned u = __float_as_uint(f);
  u = (u + 0x7fffu + ((u >> 16) & 1u)) >> 16;   // round-to-nearest-even
  return (unsigned short)u;
}

// ---------------------------------------------------------------------------
// Kernel 1: convert Wq|Wk|Wv fp32 -> bf16 into workspace (row-major [n][k])
// ---------------------------------------------------------------------------
__global__ void wconvert_kernel(const float* __restrict__ Wq,
                                const float* __restrict__ Wk,
                                const float* __restrict__ Wv,
                                unsigned short* __restrict__ dst) {
  int idx4 = blockIdx.x * blockDim.x + threadIdx.x;   // 4 floats per thread
  const int total4 = 3 * V_ * DIN / 4;
  if (idx4 >= total4) return;
  int flat = idx4 * 4;
  int p = flat >> 19;                 // 512*1024 = 2^19 elements per matrix
  int rem = flat & ((1 << 19) - 1);
  const float* src = (p == 0) ? Wq : (p == 1) ? Wk : Wv;
  float4 f = *(const float4*)(src + rem);
  ushort4 o;
  o.x = f2bf(f.x); o.y = f2bf(f.y); o.z = f2bf(f.z); o.w = f2bf(f.w);
  *(ushort4*)(dst + flat) = o;
}

// ---------------------------------------------------------------------------
// Kernel 2: x-stationary fused QKV projection + retention reduction over S.
// Grid (S/32, B), 1024 threads = 32 waves. Wave w owns output dims
// [16w, 16w+16) for q, k and v (6 WMMA accumulators = 48 VGPRs -> no spill).
// ---------------------------------------------------------------------------
__global__ __launch_bounds__(1024) void retention_main_kernel(
    const float* __restrict__ x,
    const unsigned short* __restrict__ W,   // bf16: [3][512][1024]
    float* __restrict__ acc) {              // f32:  [B][512]
  extern __shared__ unsigned short smem[]; // SCHUNK x LDSPITCH bf16

  const int tid  = threadIdx.x;
  const int wave = tid >> 5;
  const int lane = tid & 31;
  const int ln   = lane & 15;
  const int lh   = lane >> 4;
  const int b    = blockIdx.y;
  const int s0   = blockIdx.x * SCHUNK;

  // ---- stage x tile (32 x 1024) fp32 -> bf16 into LDS, coalesced float4 ----
  const float4* xg = (const float4*)(x + ((size_t)b * S_ + s0) * DIN);
  for (int i = tid; i < SCHUNK * DIN / 4; i += 1024) {
    int row = i >> 8;            // 256 float4 per row
    int c4  = i & 255;
    float4 f = xg[row * 256 + c4];
    ushort4 o;
    o.x = f2bf(f.x); o.y = f2bf(f.y); o.z = f2bf(f.z); o.w = f2bf(f.w);
    *(ushort4*)(smem + row * LDSPITCH + c4 * 4) = o;
  }
  __syncthreads();

  // ---- per-wave pointers -------------------------------------------------
  const int n0 = wave * 16;
  // B-matrix (32x16 bf16), this code's k order (matches A): lane-half lh holds 16 contiguous K values
  // for column n = lane&15  ->  read 32B from weight row (n0+ln), col k0+16*lh
  const unsigned short* wq = W + ((size_t)0 << 19) + (size_t)(n0 + ln) * DIN + 16 * lh;
  const unsigned short* wk = W + ((size_t)1 << 19) + (size_t)(n0 + ln) * DIN + 16 * lh;
  const unsigned short* wv = W + ((size_t)2 << 19) + (size_t)(n0 + ln) * DIN + 16 * lh;
  // A-matrix (16x32 bf16), this code's k order (matches B): lane holds K {kb..kb+15}, kb=16*lh
  const unsigned short* abase = smem + (size_t)ln * LDSPITCH + 16 * lh;

  v8f z = {};
  v8f qa[2], ka[2], va[2];
#pragma unroll
  for (int t = 0; t < 2; ++t) { qa[t] = z; ka[t] = z; va[t] = z; }

  // ---- K loop: 1024 in steps of 32, 6 WMMAs per step ---------------------
  for (int k0 = 0; k0 < DIN; k0 += 32) {
    AFrag a[2];
#pragma unroll
    for (int t = 0; t < 2; ++t) {
      const unsigned short* pa = abase + t * 16 * LDSPITCH + k0;
      a[t].q[0] = *(const uint4*)pa;          // K kb..kb+7
      a[t].q[1] = *(const uint4*)(pa + 8);    // K kb+8..kb+15
    }
    AFrag bq, bk, bv;
    bq.q[0] = *(const uint4*)(wq + k0); bq.q[1] = *(const uint4*)(wq + k0 + 8);
    bk.q[0] = *(const uint4*)(wk + k0); bk.q[1] = *(const uint4*)(wk + k0 + 8);
    bv.q[0] = *(const uint4*)(wv + k0); bv.q[1] = *(const uint4*)(wv + k0 + 8);
#pragma unroll
    for (int t = 0; t < 2; ++t) {
      qa[t] = __builtin_amdgcn_wmma_f32_16x16x32_bf16(false, a[t].v, false, bq.v,
                                                      (short)0, qa[t], false, false);
      ka[t] = __builtin_amdgcn_wmma_f32_16x16x32_bf16(false, a[t].v, false, bk.v,
                                                      (short)0, ka[t], false, false);
      va[t] = __builtin_amdgcn_wmma_f32_16x16x32_bf16(false, a[t].v, false, bv.v,
                                                      (short)0, va[t], false, false);
    }
  }

  // ---- retention epilogue: out[d] += sum_s q[d^1]*k[d^1]*v[d] ------------
  // C tile: lane L holds col N = L&15, rows M = g + 8*(L>>4).  d^1 swap of
  // adjacent columns == shfl_xor lane 1; row-half reduction == shfl_xor 16.
  float red = 0.f;
#pragma unroll
  for (int t = 0; t < 2; ++t) {
#pragma unroll
    for (int g = 0; g < 8; ++g) {
      float p  = qa[t][g] * ka[t][g];
      float ps = __shfl_xor(p, 1, 32);
      red = fmaf(ps, va[t][g], red);
    }
  }
  red += __shfl_xor(red, 16, 32);
  if (lane < 16) atomicAdd(&acc[b * V_ + n0 + ln], red);
}

// ---------------------------------------------------------------------------
// Kernel 3: scale -> RMSNorm -> exact GELU -> (120x512) output GEMV
// ---------------------------------------------------------------------------
__global__ void retention_final_kernel(const float* __restrict__ acc,
                                       const float* __restrict__ gamma,
                                       const float* __restrict__ Wout,
                                       float* __restrict__ out) {
  const int b = blockIdx.x, tid = threadIdx.x;   // 128 threads
  __shared__ float g[V_];
  __shared__ float red[128];
  const float cfac = 0.1f / ((float)HD_ * (float)S_);  // 0.1 * scaling^2 / S

  float ov[4];
  float local = 0.f;
#pragma unroll
  for (int i = 0; i < 4; ++i) {
    float v = acc[b * V_ + tid + i * 128] * cfac;
    ov[i] = v;
    local += v * v;
  }
  red[tid] = local;
  __syncthreads();
  for (int s = 64; s > 0; s >>= 1) {
    if (tid < s) red[tid] += red[tid + s];
    __syncthreads();
  }
  float r = rsqrtf(red[0] / (float)V_ + 1e-5f);
#pragma unroll
  for (int i = 0; i < 4; ++i) {
    int idx = tid + i * 128;
    float t = ov[i] * r * gamma[idx];
    g[idx] = 0.5f * t * (1.0f + erff(t * 0.70710678118654752f));
  }
  __syncthreads();
  if (tid < FOUT) {
    float d = 0.f;
    const float* wrow = Wout + tid * V_;
    for (int v = 0; v < V_; ++v) d = fmaf(g[v], wrow[v], d);
    out[b * FOUT + tid] = d;
  }
}

// ---------------------------------------------------------------------------
extern "C" void kernel_launch(void* const* d_in, const int* in_sizes, int n_in,
                              void* d_out, int out_size, void* d_ws, size_t ws_size,
                              hipStream_t stream) {
  const float* x     = (const float*)d_in[0];
  const float* Wq    = (const float*)d_in[1];
  const float* Wk    = (const float*)d_in[2];
  const float* Wv    = (const float*)d_in[3];
  const float* Wout  = (const float*)d_in[4];
  const float* gamma = (const float*)d_in[5];

  float* acc = (float*)d_ws;                                    // 16*512 f32
  unsigned short* Wbf = (unsigned short*)((char*)d_ws + 32768); // 3*512*1024 bf16

  hipMemsetAsync(acc, 0, B_ * V_ * sizeof(float), stream);

  wconvert_kernel<<<(3 * V_ * DIN / 4 + 255) / 256, 256, 0, stream>>>(Wq, Wk, Wv, Wbf);

  retention_main_kernel<<<dim3(S_ / SCHUNK, B_), 1024,
                          SCHUNK * LDSPITCH * sizeof(unsigned short), stream>>>(x, Wbf, acc);

  retention_final_kernel<<<B_, 128, 0, stream>>>(acc, gamma, Wout, (float*)d_out);
}
